// PointNetSetAbstractionMsg_15496242004849
// MI455X (gfx1250) — matrix operand order repaired and checked by reading, not compiled, not run on hardware
//
#include <hip/hip_runtime.h>
#include <hip/hip_bf16.h>

// ---------------------------------------------------------------------------
// PointNet++ Set Abstraction (MSG) for MI455X / gfx1250
//   FPS -> ball query (ballot-based, index-ordered) -> group+pad(bf16)
//   -> per layer: [bn+relu fused A-load] bf16 WMMA GEMM (f32 accum),
//      register-tiled 16x64 per wave (1 A-frag feeds 4 WMMAs)
//   -> deterministic 2-stage batch stats -> bn params -> final bn+relu+maxpool
// ---------------------------------------------------------------------------

#define B_      8
#define N_      4096
#define NPOINT_ 512
#define NCEN_   (B_ * NPOINT_)   // 4096 centroids
#define INCH_   64
#define CIN0_   67               // 64 point feats + 3 xyz
#define CIN0P_  96               // padded to multiple of 32
#define MMAX_   ((size_t)NCEN_ * 128)
#define BN_EPS_ 1e-5f

typedef __attribute__((ext_vector_type(16))) __bf16 v16bf;
typedef __attribute__((ext_vector_type(8)))  float  v8f;

struct __align__(16) Bf8 { __bf16 e[8]; };   // 16-byte vector of 8 bf16

// ---------------------------------------------------------------------------
// 1. Farthest point sampling: one block per batch, dist cached in LDS,
//    512 sequential argmax steps with first-index tie-break (matches argmax).
// ---------------------------------------------------------------------------
__global__ void fps_kernel(const float* __restrict__ xyz,   // (B,3,N)
                           float* __restrict__ new_xyz,     // ws (NCEN,3)
                           float* __restrict__ out_xyz)     // d_out (B,3,NPOINT)
{
    __shared__ float dist[N_];
    __shared__ float rv[256];
    __shared__ int   ri[256];

    const int b = blockIdx.x;
    const int t = threadIdx.x;
    const float* xb = xyz + (size_t)b * 3 * N_;

    for (int i = t; i < N_; i += 256) dist[i] = 1e10f;
    __syncthreads();

    int far = 0;
    for (int it = 0; it < NPOINT_; ++it) {
        const float cx = xb[0 * N_ + far];
        const float cy = xb[1 * N_ + far];
        const float cz = xb[2 * N_ + far];
        if (t == 0) {
            const int cen = b * NPOINT_ + it;
            new_xyz[cen * 3 + 0] = cx;
            new_xyz[cen * 3 + 1] = cy;
            new_xyz[cen * 3 + 2] = cz;
            out_xyz[((size_t)b * 3 + 0) * NPOINT_ + it] = cx;
            out_xyz[((size_t)b * 3 + 1) * NPOINT_ + it] = cy;
            out_xyz[((size_t)b * 3 + 2) * NPOINT_ + it] = cz;
        }
        float bv = -1.0f; int bi = 0;
        for (int i = t; i < N_; i += 256) {
            const float dx = xb[0 * N_ + i] - cx;
            const float dy = xb[1 * N_ + i] - cy;
            const float dz = xb[2 * N_ + i] - cz;
            const float d  = dx * dx + dy * dy + dz * dz;
            float dd = dist[i];
            if (d < dd) { dd = d; dist[i] = dd; }
            if (dd > bv) { bv = dd; bi = i; }
        }
        rv[t] = bv; ri[t] = bi;
        __syncthreads();
        for (int s = 128; s > 0; s >>= 1) {
            if (t < s) {
                const float ov = rv[t + s]; const int oi = ri[t + s];
                if (ov > rv[t] || (ov == rv[t] && oi < ri[t])) { rv[t] = ov; ri[t] = oi; }
            }
            __syncthreads();
        }
        far = ri[0];
        __syncthreads();
    }
}

// ---------------------------------------------------------------------------
// 2. Ball query: one wave32 per centroid; ballot + popcount prefix gives
//    first-K qualifying indices in ascending order; pad with first hit.
// ---------------------------------------------------------------------------
__device__ __forceinline__ unsigned ballot32(bool p) {
#if __has_builtin(__builtin_amdgcn_ballot_w32)
    return __builtin_amdgcn_ballot_w32(p);
#else
    return (unsigned)__ballot(p);
#endif
}

__global__ void ballquery_kernel(const float* __restrict__ xyz,
                                 const float* __restrict__ new_xyz,
                                 int* __restrict__ idx, float r2, int K)
{
    const int tid  = blockIdx.x * blockDim.x + threadIdx.x;
    const int cen  = tid >> 5;
    const int lane = tid & 31;
    if (cen >= NCEN_) return;
    const int b = cen / NPOINT_;
    const float cx = new_xyz[cen * 3 + 0];
    const float cy = new_xyz[cen * 3 + 1];
    const float cz = new_xyz[cen * 3 + 2];
    const float* xb = xyz + (size_t)b * 3 * N_;
    int* out = idx + (size_t)cen * K;

    int cnt = 0, firstn = -1;
    for (int base = 0; base < N_ && cnt < K; base += 32) {
        const int n = base + lane;
        const float dx = xb[0 * N_ + n] - cx;
        const float dy = xb[1 * N_ + n] - cy;
        const float dz = xb[2 * N_ + n] - cz;
        const bool pred = (dx * dx + dy * dy + dz * dz) <= r2;
        const unsigned mm = ballot32(pred);
        const int before = __popc(mm & ((1u << lane) - 1u));
        const int pos = cnt + before;
        if (pred && pos < K) out[pos] = n;
        if (firstn < 0 && pred && pos == 0) firstn = n;
        cnt += __popc(mm);
    }
    const unsigned fm = ballot32(firstn >= 0);
    const int first = fm ? __shfl(firstn, (int)__builtin_ctz(fm), 32) : 0;
    for (int j = cnt + lane; j < K; j += 32) out[j] = first;
}

// ---------------------------------------------------------------------------
// 3. Group + concat + channel-pad -> bf16 features X0[m][96]
//    c<64: point feats; c 64..66: xyz - centroid; c>=67: 0
// ---------------------------------------------------------------------------
__global__ void group_kernel(const float* __restrict__ xyz,
                             const float* __restrict__ points,
                             const int* __restrict__ idx,
                             const float* __restrict__ new_xyz,
                             __bf16* __restrict__ X0, int K, int M)
{
    const long long tid = (long long)blockIdx.x * blockDim.x + threadIdx.x;
    if (tid >= (long long)M * CIN0P_) return;
    const int c = (int)(tid % CIN0P_);
    const int m = (int)(tid / CIN0P_);
    const int cen = m / K;
    const int b = cen / NPOINT_;
    const int p = idx[(size_t)cen * K + (m % K)];
    float v;
    if (c < INCH_)       v = points[((size_t)b * INCH_ + c) * N_ + p];
    else if (c < CIN0_)  { const int d = c - INCH_;
                           v = xyz[((size_t)b * 3 + d) * N_ + p] - new_xyz[cen * 3 + d]; }
    else                 v = 0.0f;
    X0[(size_t)m * CIN0P_ + c] = (__bf16)v;
}

// ---------------------------------------------------------------------------
// 4. Weight conversion f32 -> padded bf16 [cout][cin_pad]
// ---------------------------------------------------------------------------
__global__ void convert_w_kernel(const float* __restrict__ W, __bf16* __restrict__ Wp,
                                 int cout, int cin, int cin_pad)
{
    const int tid = blockIdx.x * blockDim.x + threadIdx.x;
    if (tid >= cout * cin_pad) return;
    const int o = tid / cin_pad, k = tid % cin_pad;
    Wp[tid] = (__bf16)((k < cin) ? W[(size_t)o * cin + k] : 0.0f);
}

// ---------------------------------------------------------------------------
// 5. GEMM: one wave per 16x64 output tile (4 accumulators). K stepped by 32;
//    each iteration: 1 shared A fragment (BN+ReLU fused when APPLY_BN) feeds
//    4x v_wmma_f32_16x16x32_bf16. Epilogue adds bias, stores bf16 (pre-BN).
// ---------------------------------------------------------------------------
template <int APPLY_BN>
__global__ void gemm_bn_kernel(const __bf16* __restrict__ Xin, int cin,
                               const __bf16* __restrict__ Wp,
                               const float* __restrict__ bias,
                               const float* __restrict__ bnscale,
                               const float* __restrict__ bnshift,
                               __bf16* __restrict__ Yout, int cout,
                               int nMt, int totalWaves)
{
    const int tid  = blockIdx.x * blockDim.x + threadIdx.x;
    const int wave = tid >> 5;
    if (wave >= totalWaves) return;
    const int lane = tid & 31;
    const int mt = wave % nMt;        // 16-row tile
    const int nt = wave / nMt;        // 64-column tile
    const int l15 = lane & 15;
    const int hi  = (lane >> 4) & 1;
    const int row    = mt * 16 + l15;           // A row (M)
    const int ncol0  = nt * 64 + l15;           // first of 4 B/D columns
    const int khalfA = hi ? 8 : 0;              // A K-offset per ISA layout
    const int koffB  = hi ? 8 : 0;              // B K-offset per ISA layout

    v8f acc[4];
#pragma unroll
    for (int j = 0; j < 4; ++j) acc[j] = (v8f){0.f,0.f,0.f,0.f,0.f,0.f,0.f,0.f};

    const __bf16* xrow = Xin + (size_t)row * cin;
    const __bf16* wrow[4];
#pragma unroll
    for (int j = 0; j < 4; ++j) wrow[j] = Wp + (size_t)(ncol0 + 16 * j) * cin;

    for (int kt = 0; kt < cin; kt += 32) {
        const Bf8 a0 = *(const Bf8*)(xrow + kt + khalfA);
        const Bf8 a1 = *(const Bf8*)(xrow + kt + 16 + khalfA);
        v16bf av;
        if (APPLY_BN) {
#pragma unroll
            for (int e = 0; e < 8; ++e) {
                const int k0 = kt + khalfA + e;
                const int k1 = kt + 16 + khalfA + e;
                const float x0 = fmaxf(fmaf((float)a0.e[e], bnscale[k0], bnshift[k0]), 0.f);
                const float x1 = fmaxf(fmaf((float)a1.e[e], bnscale[k1], bnshift[k1]), 0.f);
                av[e] = (__bf16)x0; av[e + 8] = (__bf16)x1;
            }
        } else {
#pragma unroll
            for (int e = 0; e < 8; ++e) { av[e] = a0.e[e]; av[e + 8] = a1.e[e]; }
        }
#pragma unroll
        for (int j = 0; j < 4; ++j) {
            const Bf8 b0 = *(const Bf8*)(wrow[j] + kt + koffB);
            const Bf8 b1 = *(const Bf8*)(wrow[j] + kt + 16 + koffB);
            v16bf bv;
#pragma unroll
            for (int e = 0; e < 8; ++e) { bv[e] = b0.e[e]; bv[e + 8] = b1.e[e]; }
            acc[j] = __builtin_amdgcn_wmma_f32_16x16x32_bf16(
                         false, av, false, bv, (short)0, acc[j], false, false);
        }
    }

    const int rbase = mt * 16 + (hi ? 8 : 0);   // D: VGPR g -> rows {g, g+8}
#pragma unroll
    for (int j = 0; j < 4; ++j) {
        const int nc = ncol0 + 16 * j;
        const float bb = bias[nc];
#pragma unroll
        for (int g = 0; g < 8; ++g) {
            const float v = acc[j][g] + bb;
            Yout[(size_t)(rbase + g) * cout + nc] = (__bf16)v;
        }
    }
}

// ---------------------------------------------------------------------------
// 6. Deterministic batch stats: stage 1 = 512 partial blocks, one channel per
//    thread (coalesced row-major reads); stage 2 folded into bnparam kernel.
// ---------------------------------------------------------------------------
__global__ void stats_partial_kernel(const __bf16* __restrict__ Y, int cout,
                                     int rowsPer, float* __restrict__ part, int P)
{
    const int c = threadIdx.x;      // blockDim.x == cout
    const int p = blockIdx.x;
    const __bf16* yp = Y + (size_t)p * rowsPer * cout + c;
    float s = 0.f, sq = 0.f;
    for (int r = 0; r < rowsPer; ++r) {
        const float v = (float)yp[(size_t)r * cout];
        s += v; sq += v * v;
    }
    part[(size_t)p * cout + c] = s;
    part[(size_t)P * cout + (size_t)p * cout + c] = sq;
}

__global__ void bnparam_kernel(const float* __restrict__ part, int P, int cout,
                               const float* __restrict__ gamma,
                               const float* __restrict__ beta, float invM,
                               float* __restrict__ scale, float* __restrict__ shift)
{
    const int c = threadIdx.x;
    if (c >= cout) return;
    float s = 0.f, sq = 0.f;
    for (int p = 0; p < P; ++p) {
        s  += part[(size_t)p * cout + c];
        sq += part[(size_t)P * cout + (size_t)p * cout + c];
    }
    const float mean = s * invM;
    const float var  = sq * invM - mean * mean;
    const float sc   = gamma[c] * rsqrtf(var + BN_EPS_);
    scale[c] = sc;
    shift[c] = beta[c] - mean * sc;
}

// ---------------------------------------------------------------------------
// 7. Final BN + ReLU + max over K samples -> transposed output slice
// ---------------------------------------------------------------------------
__global__ void maxpool_kernel(const __bf16* __restrict__ Y,
                               const float* __restrict__ scale,
                               const float* __restrict__ shift,
                               float* __restrict__ out, int K, int cout, int coff)
{
    const int tid = blockIdx.x * blockDim.x + threadIdx.x;
    if (tid >= NCEN_ * cout) return;
    const int c = tid % cout;
    const int cen = tid / cout;
    const int b = cen / NPOINT_, s = cen % NPOINT_;
    const __bf16* yp = Y + (size_t)cen * K * cout + c;
    const float sc = scale[c], sh = shift[c];
    float m = 0.0f;                        // ReLU outputs are >= 0
    for (int j = 0; j < K; ++j) {
        float v = fmaf((float)yp[(size_t)j * cout], sc, sh);
        v = fmaxf(v, 0.f);
        m = fmaxf(m, v);
    }
    out[((size_t)b * 640 + coff + c) * NPOINT_ + s] = m;
}

// ---------------------------------------------------------------------------
// Host launcher
// ---------------------------------------------------------------------------
static inline const float* leaf(void* const* d_in, int s, int l, int item) {
    return (const float*)d_in[2 + (s * 3 + l) * 4 + item];
}

extern "C" void kernel_launch(void* const* d_in, const int* in_sizes, int n_in,
                              void* d_out, int out_size, void* d_ws, size_t ws_size,
                              hipStream_t stream)
{
    const float* xyz    = (const float*)d_in[0];
    const float* points = (const float*)d_in[1];

    char* ws = (char*)d_ws;
    size_t off = 0;
    auto alloc = [&](size_t bytes) -> char* {
        char* p = ws + off;
        off = (off + bytes + 255) & ~(size_t)255;
        return p;
    };
    float*  new_xyz = (float*)alloc((size_t)NCEN_ * 3 * sizeof(float));
    int*    idxbuf  = (int*)  alloc((size_t)NCEN_ * 128 * sizeof(int));
    __bf16* Wp      = (__bf16*)alloc((size_t)256 * 256 * sizeof(__bf16));
    float*  part    = (float*)alloc((size_t)512 * 256 * 2 * sizeof(float));
    float*  bnscale = (float*)alloc(256 * sizeof(float));
    float*  bnshift = (float*)alloc(256 * sizeof(float));
    __bf16* X0      = (__bf16*)alloc(MMAX_ * CIN0P_ * sizeof(__bf16));
    __bf16* YA      = (__bf16*)alloc(MMAX_ * 256 * sizeof(__bf16));
    __bf16* YB      = (__bf16*)alloc(MMAX_ * 128 * sizeof(__bf16));

    float* out_xyz = (float*)d_out;                       // (B,3,NPOINT)
    float* out_pts = out_xyz + (size_t)B_ * 3 * NPOINT_;  // (B,640,NPOINT)

    fps_kernel<<<B_, 256, 0, stream>>>(xyz, new_xyz, out_xyz);

    const int   Ks[3]      = {16, 32, 128};
    const float R2s[3]     = {0.04f, 0.16f, 0.64f};
    const int   COUT[3][3] = {{64, 64, 128}, {128, 128, 256}, {128, 128, 256}};
    const int   coffs[3]   = {0, 128, 384};

    for (int s = 0; s < 3; ++s) {
        const int K = Ks[s];
        const int M = NCEN_ * K;

        ballquery_kernel<<<NCEN_ * 32 / 256, 256, 0, stream>>>(
            xyz, new_xyz, idxbuf, R2s[s], K);

        {
            const long long tot = (long long)M * CIN0P_;
            group_kernel<<<(unsigned)((tot + 255) / 256), 256, 0, stream>>>(
                xyz, points, idxbuf, new_xyz, X0, K, M);
        }

        const __bf16* Xin = X0;
        int cin_pad = CIN0P_, cin_real = CIN0_;
        __bf16* Youts[3] = {YA, YB, YA};   // ping-pong; layer3 overwrites layer1

        for (int l = 0; l < 3; ++l) {
            const int cout = COUT[s][l];
            const float* w     = leaf(d_in, s, l, 0);
            const float* bias  = leaf(d_in, s, l, 1);
            const float* gamma = leaf(d_in, s, l, 2);
            const float* beta  = leaf(d_in, s, l, 3);

            {
                const int tot = cout * cin_pad;
                convert_w_kernel<<<(tot + 255) / 256, 256, 0, stream>>>(
                    w, Wp, cout, cin_real, cin_pad);
            }

            const int nMt = M / 16;
            const int totalWaves = nMt * (cout / 64);   // 16x64 tile per wave
            if (l == 0) {
                gemm_bn_kernel<0><<<totalWaves / 8, 256, 0, stream>>>(
                    Xin, cin_pad, Wp, bias, bnscale, bnshift,
                    Youts[l], cout, nMt, totalWaves);
            } else {
                gemm_bn_kernel<1><<<totalWaves / 8, 256, 0, stream>>>(
                    Xin, cin_pad, Wp, bias, bnscale, bnshift,
                    Youts[l], cout, nMt, totalWaves);
            }

            stats_partial_kernel<<<512, cout, 0, stream>>>(
                Youts[l], cout, M / 512, part, 512);
            bnparam_kernel<<<1, cout, 0, stream>>>(
                part, 512, cout, gamma, beta, 1.0f / (float)M, bnscale, bnshift);

            Xin = Youts[l];
            cin_pad = cout;
            cin_real = cout;
        }

        maxpool_kernel<<<(NCEN_ * COUT[s][2]) / 256, 256, 0, stream>>>(
            YA, bnscale, bnshift, out_pts, K, COUT[s][2], coffs[s]);
    }
}
